// Tacotron2Emotion_40252433498451
// MI455X (gfx1250) — matrix operand order repaired and checked by reading, not compiled, not run on hardware
//
#include <hip/hip_runtime.h>
#include <hip/hip_bf16.h>

#define B_    32
#define TIN   512
#define TOUT  800
#define NMEL  80
#define ENC   512
#define ARNN  1024
#define DRNN  1024
#define PRE   256
#define ATTD  128
#define LOCF  32
#define LOCK  31

typedef __attribute__((ext_vector_type(16))) __bf16 v16bf;
typedef __attribute__((ext_vector_type(8)))  float  v8f;
typedef __attribute__((ext_vector_type(4)))  float  f32x4;

union Frag { v16bf v; f32x4 q[2]; };

__device__ __forceinline__ float sigf(float x){ return 1.0f/(1.0f+__expf(-x)); }

__device__ __forceinline__ v8f wmma_bf16(const Frag& a, const Frag& b, v8f c){
  return __builtin_amdgcn_wmma_f32_16x16x32_bf16(false, a.v, false, b.v, (short)0, c, false, false);
}

// CDNA5 async copy: global -> LDS, 16B per lane, tracked by ASYNCcnt.
__device__ __forceinline__ void async_b128(unsigned lds_off, const __bf16* g){
  asm volatile("global_load_async_to_lds_b128 %0, %1, off"
               :: "v"(lds_off), "v"(g) : "memory");
}
__device__ __forceinline__ void wait_async0(){
  asm volatile("s_wait_asynccnt 0x0" ::: "memory");
}

// ---------------------------------------------------------------------------
// Prep: pack LSTM weights (wih|whh concat) to bf16, fold biases, pad prenet W1.
// ---------------------------------------------------------------------------
__global__ __launch_bounds__(256) void prep_pack(
    const float* __restrict__ wih_a, const float* __restrict__ whh_a,
    const float* __restrict__ bih_a, const float* __restrict__ bhh_a,
    const float* __restrict__ wih_d, const float* __restrict__ whh_d,
    const float* __restrict__ bih_d, const float* __restrict__ bhh_d,
    const float* __restrict__ w1, const float* __restrict__ w2,
    const float* __restrict__ mw,
    __bf16* __restrict__ Wa, __bf16* __restrict__ Wd,
    float* __restrict__ ba, float* __restrict__ bd,
    __bf16* __restrict__ W1p, __bf16* __restrict__ W2p, __bf16* __restrict__ Mwp)
{
  size_t i = (size_t)blockIdx.x*blockDim.x + threadIdx.x;
  size_t st = (size_t)gridDim.x*blockDim.x;
  const size_t NA = (size_t)4096*1792, ND = (size_t)4096*2560;
  for (size_t x=i; x<NA; x+=st){
    size_t n = x/1792, k = x - n*1792;
    float v = (k < 768) ? wih_a[n*768+k] : whh_a[n*1024 + (k-768)];
    Wa[x] = (__bf16)v;
  }
  for (size_t x=i; x<ND; x+=st){
    size_t n = x/2560, k = x - n*2560;
    float v = (k < 1536) ? wih_d[n*1536+k] : whh_d[n*1024 + (k-1536)];
    Wd[x] = (__bf16)v;
  }
  for (size_t x=i; x<4096; x+=st){ ba[x]=bih_a[x]+bhh_a[x]; bd[x]=bih_d[x]+bhh_d[x]; }
  for (size_t x=i; x<(size_t)256*96; x+=st){
    size_t n=x/96, k=x-n*96;
    W1p[x] = (__bf16)((k<NMEL)? w1[n*NMEL+k] : 0.f);
  }
  for (size_t x=i; x<(size_t)256*256; x+=st) W2p[x] = (__bf16)w2[x];
  for (size_t x=i; x<(size_t)128*512; x+=st) Mwp[x] = (__bf16)mw[x];
}

// ---------------------------------------------------------------------------
// Prep: prenet for all 800 steps (one WG / timestep, 16 wave32), bf16 output.
// ---------------------------------------------------------------------------
__global__ __launch_bounds__(512) void prenet_kernel(
    const float* __restrict__ dec_inputs, const __bf16* __restrict__ W1p,
    const __bf16* __restrict__ W2p, __bf16* __restrict__ pren_bf)
{
  __shared__ __align__(16) __bf16 Ain[32][104];
  __shared__ __align__(16) __bf16 X1[32][264];
  const int t = blockIdx.x, tid = threadIdx.x;
  for (int idx = tid; idx < 32*96; idx += 512){
    int m = idx/96, k = idx - m*96;
    float v = 0.f;
    if (t > 0 && k < NMEL) v = dec_inputs[((size_t)m*NMEL + k)*TOUT + (t-1)];
    Ain[m][k] = (__bf16)v;
  }
  __syncthreads();
  const int lane = tid & 31, wave = tid >> 5;
  const int n = wave*16 + (lane & 15);
  const int akb = (lane<16)?0:8, bkb=(lane<16)?0:8;
  const int mo  = (lane<16)?0:8;
  v8f a0c = {0,0,0,0,0,0,0,0}, a1c = a0c;
  for (int kc = 0; kc < 96; kc += 32){
    Frag a0,a1,bm;
    const __bf16* r0 = &Ain[lane&15][kc+akb];
    a0.q[0] = *(const f32x4*)r0;  a0.q[1] = *(const f32x4*)(r0+16);
    const __bf16* r1 = &Ain[16+(lane&15)][kc+akb];
    a1.q[0] = *(const f32x4*)r1;  a1.q[1] = *(const f32x4*)(r1+16);
    const __bf16* wb = W1p + (size_t)n*96 + kc + bkb;
    bm.q[0] = *(const f32x4*)wb;  bm.q[1] = *(const f32x4*)(wb+16);
    a0c = wmma_bf16(a0,bm,a0c);   a1c = wmma_bf16(a1,bm,a1c);
  }
  #pragma unroll
  for (int r=0;r<8;++r){
    X1[mo+r][n]    = (__bf16)fmaxf(a0c[r], 0.f);
    X1[16+mo+r][n] = (__bf16)fmaxf(a1c[r], 0.f);
  }
  __syncthreads();
  v8f b0c = {0,0,0,0,0,0,0,0}, b1c = b0c;
  for (int kc = 0; kc < 256; kc += 32){
    Frag a0,a1,bm;
    const __bf16* r0 = &X1[lane&15][kc+akb];
    a0.q[0] = *(const f32x4*)r0;  a0.q[1] = *(const f32x4*)(r0+16);
    const __bf16* r1 = &X1[16+(lane&15)][kc+akb];
    a1.q[0] = *(const f32x4*)r1;  a1.q[1] = *(const f32x4*)(r1+16);
    const __bf16* wb = W2p + (size_t)n*256 + kc + bkb;
    bm.q[0] = *(const f32x4*)wb;  bm.q[1] = *(const f32x4*)(wb+16);
    b0c = wmma_bf16(a0,bm,b0c);   b1c = wmma_bf16(a1,bm,b1c);
  }
  __bf16* po = pren_bf + (size_t)t*32*PRE;
  #pragma unroll
  for (int r=0;r<8;++r){
    po[(size_t)(mo+r)*PRE + n]    = (__bf16)fmaxf(b0c[r], 0.f);
    po[(size_t)(16+mo+r)*PRE + n] = (__bf16)fmaxf(b1c[r], 0.f);
  }
}

// ---------------------------------------------------------------------------
// Prep: pmem = memory @ mem_w.T   [B*TIN, 512] x [512, 128] -> fp32.
// ---------------------------------------------------------------------------
__global__ __launch_bounds__(256) void pmem_kernel(
    const float* __restrict__ memory, const __bf16* __restrict__ Mwp,
    float* __restrict__ pmem)
{
  __shared__ __align__(16) __bf16 Ms[32][520];
  const int tid = threadIdx.x;
  const size_t r0 = (size_t)blockIdx.x * 32;
  for (int idx = tid; idx < 32*512; idx += 256){
    int m = idx >> 9, k = idx & 511;
    Ms[m][k] = (__bf16)memory[(r0+m)*ENC + k];
  }
  __syncthreads();
  const int lane = tid & 31, wave = tid >> 5;
  const int n = wave*16 + (lane & 15);
  const int akb = (lane<16)?0:8, bkb=(lane<16)?0:8;
  const int mo  = (lane<16)?0:8;
  v8f c0 = {0,0,0,0,0,0,0,0}, c1 = c0;
  for (int kc = 0; kc < 512; kc += 32){
    Frag a0,a1,bm;
    const __bf16* p0 = &Ms[lane&15][kc+akb];
    a0.q[0] = *(const f32x4*)p0;  a0.q[1] = *(const f32x4*)(p0+16);
    const __bf16* p1 = &Ms[16+(lane&15)][kc+akb];
    a1.q[0] = *(const f32x4*)p1;  a1.q[1] = *(const f32x4*)(p1+16);
    const __bf16* wb = Mwp + (size_t)n*512 + kc + bkb;
    bm.q[0] = *(const f32x4*)wb;  bm.q[1] = *(const f32x4*)(wb+16);
    c0 = wmma_bf16(a0,bm,c0);     c1 = wmma_bf16(a1,bm,c1);
  }
  #pragma unroll
  for (int r=0;r<8;++r){
    pmem[(r0 + mo + r)*ATTD + n]      = c0[r];
    pmem[(r0 + 16 + mo + r)*ATTD + n] = c1[r];
  }
}

// ---------------------------------------------------------------------------
// Persistent decoder: 32 WGs x 256 threads, grid-synced atomic barrier.
// ---------------------------------------------------------------------------
__device__ void gsync(unsigned* cnt){
  __threadfence();
  __syncthreads();
  if (threadIdx.x == 0){
    unsigned prev = atomicAdd(cnt, 1u);
    unsigned tgt  = (prev/32u + 1u)*32u;
    while (atomicAdd(cnt, 0u) < tgt) __builtin_amdgcn_s_sleep(1);
  }
  __syncthreads();
  __threadfence();
}

// z[32x4096] = X[32xK] @ W[4096xK].T + bias ; X gathered from up to 3 bf16
// sources, async-copied into LDS in 512-wide K blocks.
__device__ void gemm_stage(const __bf16* __restrict__ W, const float* __restrict__ bias,
                           float* __restrict__ z,
                           const __bf16* x0, int l0, int ld0,
                           const __bf16* x1, int l1, int ld1,
                           const __bf16* x2, int ld2,
                           int Ktot, __bf16 (*xs)[520], int tid)
{
  const int lane = tid & 31;
  const int nt   = blockIdx.x*8 + (tid>>5);     // 0..255
  const int n    = nt*16 + (lane & 15);
  const int akb  = (lane < 16) ? 0 : 8;
  const int bkb  = (lane < 16) ? 0 : 8;
  const unsigned xs_base = (unsigned)(size_t)(&xs[0][0]);
  v8f acc0 = {0,0,0,0,0,0,0,0}, acc1 = acc0;
  const __bf16* wrow = W + (size_t)n * (size_t)Ktot;
  for (int kb0 = 0; kb0 < Ktot; kb0 += 512){
    const int KB = (Ktot - kb0 < 512) ? (Ktot - kb0) : 512;
    const int per_row = KB >> 3;                // 16B async ops per row
    for (int idx = tid; idx < 32*per_row; idx += 256){
      int m = idx / per_row, k = (idx - m*per_row) << 3;
      int gk = kb0 + k;
      const __bf16* src;
      if (gk < l0)            src = x0 + m*ld0 + gk;
      else if (gk < l0 + l1)  src = x1 + m*ld1 + (gk - l0);
      else                    src = x2 + m*ld2 + (gk - l0 - l1);
      async_b128(xs_base + (unsigned)(m*1040 + k*2), src);
    }
    wait_async0();
    __syncthreads();
    if (kb0 + 512 < Ktot) __builtin_prefetch(wrow + kb0 + 512, 0, 1);
    for (int kc = 0; kc < KB; kc += 32){
      Frag a0,a1,bm;
      const __bf16* r0 = &xs[lane & 15][kc + akb];
      a0.q[0] = *(const f32x4*)r0;  a0.q[1] = *(const f32x4*)(r0+16);
      const __bf16* r1 = &xs[16 + (lane & 15)][kc + akb];
      a1.q[0] = *(const f32x4*)r1;  a1.q[1] = *(const f32x4*)(r1+16);
      const __bf16* wb = wrow + kb0 + kc + bkb;
      bm.q[0] = *(const f32x4*)wb;  bm.q[1] = *(const f32x4*)(wb+16);
      acc0 = wmma_bf16(a0, bm, acc0);
      acc1 = wmma_bf16(a1, bm, acc1);
    }
    __syncthreads();
  }
  const float bv = bias[n];
  const int mo = (lane < 16) ? 0 : 8;
  #pragma unroll
  for (int r = 0; r < 8; ++r){
    z[(size_t)(mo + r)      * 4096 + n] = acc0[r] + bv;
    z[(size_t)(16 + mo + r) * 4096 + n] = acc1[r] + bv;
  }
}

__global__ __launch_bounds__(256,1) void taco_decoder(
    const float* __restrict__ memory, const __bf16* __restrict__ pren_bf,
    const float* __restrict__ pmem,
    const __bf16* __restrict__ Wa, const float* __restrict__ ba,
    const __bf16* __restrict__ Wd, const float* __restrict__ bd,
    const float* __restrict__ q_w, const float* __restrict__ loc_conv,
    const float* __restrict__ loc_lin, const float* __restrict__ v_w,
    const float* __restrict__ proj_w, const float* __restrict__ proj_b,
    const float* __restrict__ gate_w, const float* __restrict__ gate_b,
    const int* __restrict__ mem_len,
    float* z_att, float* z_dec,
    float* ac, float* dc,
    __bf16* ah_bf, __bf16* dh_bf, __bf16* actx_bf,
    float* aw, float* awc, float* actx,
    float* out_mel, float* out_gate, float* out_align,
    unsigned* cnt)
{
  __shared__ __align__(16) union {
    __bf16 xs[32][520];
    struct {
      float e[512]; float red[256]; float hls[1024]; float pq[128];
      __bf16 locA[512][40];             // conv features, WMMA A-matrix (K=32)
    } att;
  } sm;
  __shared__ float  convw[LOCF*2*LOCK];
  __shared__ __align__(16) __bf16 llb[ATTD*LOCF];   // loc_lin as WMMA B-matrix
  __shared__ float  vv[ATTD];

  const int tid = threadIdx.x;
  const int b   = blockIdx.x;                // WG b <-> batch b in stages B/D
  for (int j=tid; j<LOCF*2*LOCK; j+=256) convw[j]=loc_conv[j];
  for (int j=tid; j<ATTD*LOCF;   j+=256) llb[j]=(__bf16)loc_lin[j];
  if (tid < ATTD) vv[tid] = v_w[tid];
  __syncthreads();

  for (int step = 0; step < TOUT; ++step){
    // ---- Stage A: att LSTM GEMM  z_att = [pren_t|actx|ah] @ Wa.T + ba
    gemm_stage(Wa, ba, z_att,
               pren_bf + (size_t)step*B_*PRE, PRE, PRE,
               actx_bf, ENC, ENC,
               ah_bf, ARNN, 1792, sm.xs, tid);
    gsync(cnt);

    // ---- Stage B (per-batch): gates -> pq -> attention -> context
    {
      const float* zb = z_att + (size_t)b*4096;
      float* acb = ac + b*ARNN;
      __bf16* ahb = ah_bf + b*ARNN;
      for (int j = tid; j < ARNN; j += 256){
        float zi = zb[j], zf = zb[ARNN+j], zg = zb[2*ARNN+j], zo = zb[3*ARNN+j];
        float cc = sigf(zf)*acb[j] + sigf(zi)*tanhf(zg);
        float hh = sigf(zo)*tanhf(cc);
        acb[j] = cc; ahb[j] = (__bf16)hh; sm.att.hls[j] = hh;
      }
      __syncthreads();
      if (tid < ATTD){
        const float* qr = q_w + (size_t)tid*ARNN;
        float s = 0.f;
        for (int k = 0; k < ARNN; ++k) s += qr[k]*sm.att.hls[k];
        sm.att.pq[tid] = s;
      }
      float* awb = aw + b*TIN; float* awcb = awc + b*TIN;
      const int len = mem_len[b];
      // location conv -> bf16 A-matrix [512 x 32] in LDS
      for (int t = tid; t < TIN; t += 256){
        float lf[LOCF];
        #pragma unroll
        for (int f = 0; f < LOCF; ++f) lf[f] = 0.f;
        for (int k = 0; k < LOCK; ++k){
          int src = t + k - 15;
          float va = 0.f, vc = 0.f;
          if (src >= 0 && src < TIN){ va = awb[src]; vc = awcb[src]; }
          #pragma unroll
          for (int f = 0; f < LOCF; ++f)
            lf[f] += convw[f*62 + k]*va + convw[f*62 + 31 + k]*vc;
        }
        #pragma unroll
        for (int f = 0; f < LOCF; ++f) sm.att.locA[t][f] = (__bf16)lf[f];
      }
      __syncthreads();
      // energies via WMMA: [512x32] @ [32x128], fused tanh + v-dot on fragments
      {
        const int lane = tid & 31, wave = tid >> 5;
        const int akb = (lane<16)?0:8, bkb=(lane<16)?0:8;
        const int mo  = (lane<16)?0:8;
        for (int mt = wave*4; mt < wave*4 + 4; ++mt){
          Frag a0;
          const __bf16* r0 = &sm.att.locA[mt*16 + (lane & 15)][akb];
          a0.q[0] = *(const f32x4*)r0;  a0.q[1] = *(const f32x4*)(r0+16);
          float esum[8];
          #pragma unroll
          for (int r=0;r<8;++r) esum[r] = 0.f;
          for (int ntl = 0; ntl < 8; ++ntl){
            Frag bm;
            const __bf16* wb = &llb[(ntl*16 + (lane & 15))*LOCF + bkb];
            bm.q[0] = *(const f32x4*)wb;  bm.q[1] = *(const f32x4*)(wb+16);
            v8f z8 = {0,0,0,0,0,0,0,0};
            z8 = wmma_bf16(a0, bm, z8);
            const int a  = ntl*16 + (lane & 15);
            const float pqa = sm.att.pq[a];
            const float vva = vv[a];
            const float* pmb = pmem + ((size_t)b*TIN + mt*16 + mo)*ATTD + a;
            #pragma unroll
            for (int r=0;r<8;++r)
              esum[r] += vva * tanhf(pqa + z8[r] + pmb[(size_t)r*ATTD]);
          }
          #pragma unroll
          for (int r=0;r<8;++r){
            float s = esum[r];
            s += __shfl_xor(s, 1, 32);
            s += __shfl_xor(s, 2, 32);
            s += __shfl_xor(s, 4, 32);
            s += __shfl_xor(s, 8, 32);
            const int t = mt*16 + mo + r;
            if ((lane & 15) == 0) sm.att.e[t] = (t < len) ? s : -1e9f;
          }
        }
      }
      __syncthreads();
      float mx = -3.4e38f;
      for (int t = tid; t < TIN; t += 256) mx = fmaxf(mx, sm.att.e[t]);
      sm.att.red[tid] = mx; __syncthreads();
      for (int off = 128; off > 0; off >>= 1){
        if (tid < off) sm.att.red[tid] = fmaxf(sm.att.red[tid], sm.att.red[tid+off]);
        __syncthreads();
      }
      mx = sm.att.red[0]; __syncthreads();
      float s = 0.f;
      for (int t = tid; t < TIN; t += 256){
        float p = __expf(sm.att.e[t]-mx); sm.att.e[t]=p; s+=p;
      }
      sm.att.red[tid] = s; __syncthreads();
      for (int off = 128; off > 0; off >>= 1){
        if (tid < off) sm.att.red[tid] += sm.att.red[tid+off];
        __syncthreads();
      }
      const float inv = 1.0f / sm.att.red[0];
      __syncthreads();
      for (int t = tid; t < TIN; t += 256){
        float p = sm.att.e[t]*inv;
        sm.att.e[t] = p;
        awb[t] = p;
        awcb[t] += p;
        out_align[((size_t)b*TOUT + step)*TIN + t] = p;
      }
      __syncthreads();
      for (int d = tid; d < ENC; d += 256){
        const float* mb = memory + ((size_t)b*TIN)*ENC + d;
        float s2 = 0.f;
        for (int t = 0; t < TIN; ++t) s2 += sm.att.e[t]*mb[(size_t)t*ENC];
        actx[b*ENC + d] = s2;
        actx_bf[b*ENC + d] = (__bf16)s2;
      }
    }
    gsync(cnt);

    // ---- Stage C: dec LSTM GEMM  z_dec = [ah|actx|dh] @ Wd.T + bd
    gemm_stage(Wd, bd, z_dec,
               ah_bf, ARNN, ARNN,
               actx_bf, ENC, ENC,
               dh_bf, DRNN, 2560, sm.xs, tid);
    gsync(cnt);

    // ---- Stage D (per-batch): gates -> projection / gate outputs
    {
      const float* zb = z_dec + (size_t)b*4096;
      float* dcb = dc + b*DRNN;
      __bf16* dhb = dh_bf + b*DRNN;
      for (int j = tid; j < DRNN; j += 256){
        float zi = zb[j], zf = zb[DRNN+j], zg = zb[2*DRNN+j], zo = zb[3*DRNN+j];
        float cc = sigf(zf)*dcb[j] + sigf(zi)*tanhf(zg);
        float hh = sigf(zo)*tanhf(cc);
        dcb[j] = cc; dhb[j] = (__bf16)hh; sm.att.hls[j] = hh;
      }
      __syncthreads();
      if (tid <= NMEL){
        const float* w = (tid < NMEL) ? (proj_w + (size_t)tid*(DRNN+ENC)) : gate_w;
        float acc = (tid < NMEL) ? proj_b[tid] : gate_b[0];
        for (int k = 0; k < DRNN; ++k) acc += w[k]*sm.att.hls[k];
        const float* ax = actx + b*ENC;
        for (int k = 0; k < ENC; ++k) acc += w[DRNN + k]*ax[k];
        if (tid < NMEL) out_mel[((size_t)b*NMEL + tid)*TOUT + step] = acc;
        else            out_gate[(size_t)b*TOUT + step] = acc;
      }
    }
    gsync(cnt);
  }
}

// ---------------------------------------------------------------------------
extern "C" void kernel_launch(void* const* d_in, const int* in_sizes, int n_in,
                              void* d_out, int out_size, void* d_ws, size_t ws_size,
                              hipStream_t stream)
{
  (void)in_sizes; (void)n_in; (void)out_size; (void)ws_size;
  const float* memory     = (const float*)d_in[0];
  const float* dec_inputs = (const float*)d_in[1];
  const float* prenet_w1  = (const float*)d_in[2];
  const float* prenet_w2  = (const float*)d_in[3];
  const float* att_wih    = (const float*)d_in[4];
  const float* att_whh    = (const float*)d_in[5];
  const float* att_bih    = (const float*)d_in[6];
  const float* att_bhh    = (const float*)d_in[7];
  const float* q_w        = (const float*)d_in[8];
  const float* mem_w      = (const float*)d_in[9];
  const float* v_w        = (const float*)d_in[10];
  const float* loc_conv_w = (const float*)d_in[11];
  const float* loc_lin_w  = (const float*)d_in[12];
  const float* dec_wih    = (const float*)d_in[13];
  const float* dec_whh    = (const float*)d_in[14];
  const float* dec_bih    = (const float*)d_in[15];
  const float* dec_bhh    = (const float*)d_in[16];
  const float* proj_w     = (const float*)d_in[17];
  const float* proj_b     = (const float*)d_in[18];
  const float* gate_w     = (const float*)d_in[19];
  const float* gate_b     = (const float*)d_in[20];
  const int*   mem_len    = (const int*)d_in[21];

  char* ws = (char*)d_ws;
  size_t off = 0;
  auto take = [&](size_t bytes)->size_t{
    size_t r = off; off += (bytes + 255) & ~(size_t)255; return r;
  };
  // ---- zero-initialized state region (memset below) ----
  size_t o_cnt    = take(4);
  size_t o_ac     = take((size_t)B_*ARNN*4);
  size_t o_dc     = take((size_t)B_*DRNN*4);
  size_t o_aw     = take((size_t)B_*TIN*4);
  size_t o_awc    = take((size_t)B_*TIN*4);
  size_t o_actx   = take((size_t)B_*ENC*4);
  size_t o_ahbf   = take((size_t)B_*ARNN*2);
  size_t o_dhbf   = take((size_t)B_*DRNN*2);
  size_t o_actxbf = take((size_t)B_*ENC*2);
  size_t state_end = off;
  // ---- fully-rewritten-per-launch scratch ----
  size_t o_Wa   = take((size_t)4096*1792*2);
  size_t o_Wd   = take((size_t)4096*2560*2);
  size_t o_ba   = take(4096*4);
  size_t o_bd   = take(4096*4);
  size_t o_W1p  = take((size_t)256*96*2);
  size_t o_W2p  = take((size_t)256*256*2);
  size_t o_Mwp  = take((size_t)128*512*2);
  size_t o_pren = take((size_t)TOUT*B_*PRE*2);
  size_t o_pmem = take((size_t)B_*TIN*ATTD*4);
  size_t o_za   = take((size_t)B_*4096*4);
  size_t o_zd   = take((size_t)B_*4096*4);

  hipMemsetAsync(ws, 0, state_end, stream);

  prep_pack<<<2048, 256, 0, stream>>>(
      att_wih, att_whh, att_bih, att_bhh,
      dec_wih, dec_whh, dec_bih, dec_bhh,
      prenet_w1, prenet_w2, mem_w,
      (__bf16*)(ws+o_Wa), (__bf16*)(ws+o_Wd),
      (float*)(ws+o_ba), (float*)(ws+o_bd),
      (__bf16*)(ws+o_W1p), (__bf16*)(ws+o_W2p), (__bf16*)(ws+o_Mwp));

  prenet_kernel<<<TOUT, 512, 0, stream>>>(
      dec_inputs, (const __bf16*)(ws+o_W1p), (const __bf16*)(ws+o_W2p),
      (__bf16*)(ws+o_pren));

  pmem_kernel<<<(B_*TIN)/32, 256, 0, stream>>>(
      memory, (const __bf16*)(ws+o_Mwp), (float*)(ws+o_pmem));

  float* out       = (float*)d_out;
  float* out_mel   = out;
  float* out_gate  = out + (size_t)B_*NMEL*TOUT;
  float* out_align = out_gate + (size_t)B_*TOUT;

  taco_decoder<<<32, 256, 0, stream>>>(
      memory, (const __bf16*)(ws+o_pren), (const float*)(ws+o_pmem),
      (const __bf16*)(ws+o_Wa), (const float*)(ws+o_ba),
      (const __bf16*)(ws+o_Wd), (const float*)(ws+o_bd),
      q_w, loc_conv_w, loc_lin_w, v_w, proj_w, proj_b, gate_w, gate_b, mem_len,
      (float*)(ws+o_za), (float*)(ws+o_zd),
      (float*)(ws+o_ac), (float*)(ws+o_dc),
      (__bf16*)(ws+o_ahbf), (__bf16*)(ws+o_dhbf), (__bf16*)(ws+o_actxbf),
      (float*)(ws+o_aw), (float*)(ws+o_awc), (float*)(ws+o_actx),
      out_mel, out_gate, out_align, (unsigned*)(ws+o_cnt));
}
